// DGCNN_20461224198152
// MI455X (gfx1250) — matrix operand order repaired and checked by reading, not compiled, not run on hardware
//
#include <hip/hip_runtime.h>
#include <hip/hip_bf16.h>

// ---------------------------------------------------------------------------
// DGCNN forward for MI455X (gfx1250). GEMM-shaped work (edge-convs, conv5)
// runs on v_wmma_f32_16x16x32_f16; irregular kNN top-20 runs on VALU.
// ---------------------------------------------------------------------------

typedef _Float16 half_t;
typedef __attribute__((ext_vector_type(2)))  _Float16 h2;
typedef __attribute__((ext_vector_type(16))) _Float16 v16h;
typedef __attribute__((ext_vector_type(8)))  float    v8f;

#define BN_SCALE 0.9999950000374997f   // 1/sqrt(1+1e-5)
#define NEG_BIG  (-3.402823466e38f)
#define KNN 20

__device__ __forceinline__ float lrelu(float v) { return v >= 0.f ? v : 0.2f * v; }

// feature accessor: bn9=1 -> layout [B,N,C] (raw input), else [B,C,N]
__device__ __forceinline__ float feat_ld(const float* __restrict__ f, int b, int c,
                                         int n, int C, int N, int bn9) {
  return bn9 ? f[(b * N + n) * C + c] : f[(b * C + c) * N + n];
}

__device__ __forceinline__ void atomicMaxF(float* addr, float val) {
  unsigned int* ua = (unsigned int*)addr;
  unsigned int cur = __float_as_uint(*addr);
  while (__uint_as_float(cur) < val) {
    unsigned int prev = atomicCAS(ua, cur, __float_as_uint(val));
    if (prev == cur) break;
    cur = prev;
  }
}

// ---------------------------------------------------------------------------
// 1) kNN top-20 (negative squared distance, self included — matches reference)
//    block = 64 threads (2 waves), each thread owns one point.
// ---------------------------------------------------------------------------
__global__ __launch_bounds__(64)
void knn_topk(const float* __restrict__ f, int* __restrict__ idx_out,
              int B, int C, int N, int bn9) {
  const int tid = threadIdx.x;
  const int n0  = blockIdx.x * 64;
  const int b   = blockIdx.y;
  const int st  = C + 1;                       // LDS row stride (bank-conflict pad)
  __shared__ float lO[64 * 129];               // this block's 64 points
  __shared__ float lT[32 * 129];               // streamed tile of candidate points

  for (int e = tid; e < 64 * C; e += 64) {
    int n = e / C, c = e - n * C;
    lO[n * st + c] = feat_ld(f, b, c, n0 + n, C, N, bn9);
  }

  float bestd[KNN];
  int   besti[KNN];
#pragma unroll
  for (int i = 0; i < KNN; ++i) { bestd[i] = NEG_BIG; besti[i] = 0; }

  for (int m0 = 0; m0 < N; m0 += 32) {
    __syncthreads();
    for (int e = tid; e < 32 * C; e += 64) {
      int mm = e / C, c = e - mm * C;
      lT[mm * st + c] = feat_ld(f, b, c, m0 + mm, C, N, bn9);
    }
    __syncthreads();
    const float* own = &lO[tid * st];
    for (int mm = 0; mm < 32; ++mm) {
      const float* oth = &lT[mm * st];
      float d = 0.f;
      for (int c = 0; c < C; ++c) {
        float t = own[c] - oth[c];
        d = fmaf(t, t, d);
      }
      float s = -d;                            // top-k of negative sq-dist
      if (s > bestd[KNN - 1]) {
        int pos = KNN - 1;
#pragma unroll
        for (int t = KNN - 1; t > 0; --t) {
          if (bestd[t - 1] < s) {
            bestd[t] = bestd[t - 1]; besti[t] = besti[t - 1];
            pos = t - 1;
          }
        }
#pragma unroll
        for (int t = 0; t < KNN; ++t)
          if (t == pos) { bestd[t] = s; besti[t] = m0 + mm; }
      }
    }
  }
#pragma unroll
  for (int i = 0; i < KNN; ++i)
    idx_out[(b * N + n0 + tid) * KNN + i] = besti[i];
}

// ---------------------------------------------------------------------------
// 2) weight f32 -> f16 with zero K-padding (Kpad = multiple of 32 >= Cin)
// ---------------------------------------------------------------------------
__global__ void cvt_w16(const float* __restrict__ src, half_t* __restrict__ dst,
                        int O, int Cin, int Kpad) {
  int i = blockIdx.x * blockDim.x + threadIdx.x;
  if (i >= O * Kpad) return;
  int o = i / Kpad, c = i - o * Kpad;
  dst[i] = (half_t)(c < Cin ? src[o * Cin + c] : 0.f);
}

__global__ void fill_neg(float* __restrict__ p, int n) {
  int i = blockIdx.x * blockDim.x + threadIdx.x;
  if (i < n) p[i] = NEG_BIG;
}

// ---------------------------------------------------------------------------
// WMMA fragment helpers (ISA §7.12.2 16-bit layouts, wave32)
//   A 16x32: lane<16 -> M=lane, K pairs 0..7 then 16..23; lane>=16 -> K 8..15/24..31
//   B 32x16: lane&15 = column N; lane<16 -> K 0..7 then 16..23; lane>=16 -> K 8..15/24..31
// ---------------------------------------------------------------------------
__device__ __forceinline__ v16h load_fragA(const half_t* __restrict__ row, int hi) {
  v16h a;
#pragma unroll
  for (int v = 0; v < 8; ++v) {
    int ka = ((v < 4) ? 0 : 16) + hi * 8 + (v & 3) * 2;
    h2 p = *(const h2*)(row + ka);
    a[2 * v] = p.x; a[2 * v + 1] = p.y;
  }
  return a;
}

__device__ __forceinline__ v16h load_fragB_lds(const half_t* row, int hi) {
  v16h bb;
#pragma unroll
  for (int v = 0; v < 8; ++v) {
    int kb = ((v < 4) ? 0 : 16) + hi * 8 + (v & 3) * 2;
    h2 p = *(const h2*)(row + kb);
    bb[2 * v] = p.x; bb[2 * v + 1] = p.y;
  }
  return bb;
}

// ---------------------------------------------------------------------------
// 3) edge-conv: out[b,o,n] = max_k lrelu(bn(sum_c W[o,c] * feat(c,n,k)))
//    block = 128 threads (4 waves); wave -> one 16-row out-channel tile,
//    block -> 16 points. k loop keeps running max in registers; no atomics.
// ---------------------------------------------------------------------------
__global__ __launch_bounds__(128)
void edgeconv_wmma(const float* __restrict__ in, const half_t* __restrict__ Wh,
                   const float* __restrict__ g, const float* __restrict__ be,
                   const int* __restrict__ idx, float* __restrict__ out,
                   int B, int C, int N, int O, int Kpad, int bn9) {
  __shared__ half_t lB[16 * 512];              // [point][Kpad] f16 panel, <=16KB
  const int b    = blockIdx.z;
  const int n0   = blockIdx.x * 16;
  const int tid  = threadIdx.x;
  const int lane = tid & 31;
  const int wave = tid >> 5;
  const int otile = blockIdx.y * 4 + wave;
  const bool active = (otile * 16) < O;
  const int hi  = lane >> 4;
  const int r16 = lane & 15;
  const int twoC = 2 * C;

  float gk[8], bk[8], vmax[8];
#pragma unroll
  for (int r = 0; r < 8; ++r) vmax[r] = NEG_BIG;
  if (active) {
#pragma unroll
    for (int r = 0; r < 8; ++r) {
      int o = otile * 16 + r + hi * 8;
      gk[r] = g[o] * BN_SCALE;
      bk[r] = be[o];
    }
  }

  for (int k = 0; k < KNN; ++k) {
    __syncthreads();
    // stage B panel: feat(c, n, k) for the 16 points, zero-padded to Kpad
    for (int e = tid; e < 16 * Kpad; e += 128) {
      int n = e / Kpad, c = e - n * Kpad;
      int gn = n0 + n;
      float v = 0.f;
      if (c < twoC) {
        if (c < C) {
          int j = idx[(b * N + gn) * KNN + k];
          v = feat_ld(in, b, c, j, C, N, bn9) - feat_ld(in, b, c, gn, C, N, bn9);
        } else {
          v = feat_ld(in, b, c - C, gn, C, N, bn9);
        }
      }
      lB[e] = (half_t)v;
    }
    __syncthreads();

    if (active) {                              // wave-uniform: EXEC all-ones at WMMA
      v8f acc = {};
      for (int ks = 0; ks < Kpad; ks += 32) {
        v16h a  = load_fragA(Wh + (otile * 16 + r16) * Kpad + ks, hi);
        v16h bb = load_fragB_lds(lB + r16 * Kpad + ks, hi);
        acc = __builtin_amdgcn_wmma_f32_16x16x32_f16(
            false, a, false, bb, (short)0, acc, false, false);
      }
#pragma unroll
      for (int r = 0; r < 8; ++r)
        vmax[r] = fmaxf(vmax[r], lrelu(acc[r] * gk[r] + bk[r]));
    }
  }

  if (active) {
#pragma unroll
    for (int r = 0; r < 8; ++r) {
      int o = otile * 16 + r + hi * 8;
      out[(b * O + o) * N + n0 + r16] = vmax[r];
    }
  }
}

// ---------------------------------------------------------------------------
// 4) conv5 (1024x512) + global max pool over N, feature input = concat(x1..x4)
// ---------------------------------------------------------------------------
__global__ __launch_bounds__(128)
void conv5_pool_wmma(const float* __restrict__ a1, const float* __restrict__ a2,
                     const float* __restrict__ a3, const float* __restrict__ a4,
                     const half_t* __restrict__ Wh, const float* __restrict__ g,
                     const float* __restrict__ be, float* __restrict__ pooled,
                     int B, int N) {
  __shared__ half_t lB[16 * 512];
  const int b    = blockIdx.z;
  const int n0   = blockIdx.x * 16;
  const int tid  = threadIdx.x;
  const int lane = tid & 31;
  const int wave = tid >> 5;
  const int otile = blockIdx.y * 4 + wave;     // 0..63, always valid
  const int hi  = lane >> 4;
  const int r16 = lane & 15;
  const int Kpad = 512;

  for (int e = tid; e < 16 * Kpad; e += 128) {
    int n = e / Kpad, c = e - n * Kpad;
    int gn = n0 + n;
    float v;
    if (c < 64)       v = a1[(b * 64  +  c       ) * N + gn];
    else if (c < 128) v = a2[(b * 64  + (c - 64 )) * N + gn];
    else if (c < 256) v = a3[(b * 128 + (c - 128)) * N + gn];
    else              v = a4[(b * 256 + (c - 256)) * N + gn];
    lB[e] = (half_t)v;
  }
  __syncthreads();

  v8f acc = {};
  for (int ks = 0; ks < Kpad; ks += 32) {
    v16h a  = load_fragA(Wh + (otile * 16 + r16) * Kpad + ks, hi);
    v16h bb = load_fragB_lds(lB + r16 * Kpad + ks, hi);
    acc = __builtin_amdgcn_wmma_f32_16x16x32_f16(
        false, a, false, bb, (short)0, acc, false, false);
  }

#pragma unroll
  for (int r = 0; r < 8; ++r) {
    int o = otile * 16 + r + hi * 8;
    float v = lrelu(acc[r] * (g[o] * BN_SCALE) + be[o]);
#pragma unroll
    for (int off = 1; off < 16; off <<= 1)     // max over the 16 columns (width 16)
      v = fmaxf(v, __shfl_xor(v, off, 16));
    if (r16 == 0) atomicMaxF(&pooled[b * 1024 + o], v);
  }
}

// ---------------------------------------------------------------------------
// 5) heads (tiny: B=8) — plain VALU dense layers
// ---------------------------------------------------------------------------
__global__ void concat_emb(const float* __restrict__ pooled,
                           const float* __restrict__ emb,
                           const int* __restrict__ obj,
                           float* __restrict__ hcat, int B) {
  int i = blockIdx.x * blockDim.x + threadIdx.x;
  if (i >= B * 1152) return;
  int b = i / 1152, c = i - b * 1152;
  hcat[i] = (c < 1024) ? pooled[b * 1024 + c] : emb[obj[b] * 128 + (c - 1024)];
}

__global__ void dense(const float* __restrict__ in, const float* __restrict__ W,
                      const float* __restrict__ bias, const float* __restrict__ g,
                      const float* __restrict__ be, float* __restrict__ out,
                      int B, int Cin, int O, int act) {
  int i = blockIdx.x * blockDim.x + threadIdx.x;
  if (i >= B * O) return;
  int b = i / O, o = i - b * O;
  const float* x = in + b * Cin;
  const float* w = W + o * Cin;
  float s = 0.f;
  for (int c = 0; c < Cin; ++c) s = fmaf(w[c], x[c], s);
  s += bias[o];
  if (act) s = lrelu(s * (g[o] * BN_SCALE) + be[o]);
  out[i] = s;
}

// ---------------------------------------------------------------------------
// launch
// ---------------------------------------------------------------------------
extern "C" void kernel_launch(void* const* d_in, const int* in_sizes, int n_in,
                              void* d_out, int out_size, void* d_ws, size_t ws_size,
                              hipStream_t stream) {
  (void)in_sizes; (void)n_in; (void)out_size; (void)ws_size;
  const int B = 8, N = 1024;

  const float* x      = (const float*)d_in[0];
  const int*   obj    = (const int*)  d_in[1];
  const float* W1 = (const float*)d_in[2],  *g1 = (const float*)d_in[3],  *be1 = (const float*)d_in[4];
  const float* W2 = (const float*)d_in[5],  *g2 = (const float*)d_in[6],  *be2 = (const float*)d_in[7];
  const float* W3 = (const float*)d_in[8],  *g3 = (const float*)d_in[9],  *be3 = (const float*)d_in[10];
  const float* W4 = (const float*)d_in[11], *g4 = (const float*)d_in[12], *be4 = (const float*)d_in[13];
  const float* W5 = (const float*)d_in[14], *g5 = (const float*)d_in[15], *be5 = (const float*)d_in[16];
  const float* emb = (const float*)d_in[17];
  const float* rW1 = (const float*)d_in[18], *rb1 = (const float*)d_in[19];
  const float* rg1 = (const float*)d_in[20], *rbe1 = (const float*)d_in[21];
  const float* rW2 = (const float*)d_in[22], *rb2 = (const float*)d_in[23];
  const float* rg2 = (const float*)d_in[24], *rbe2 = (const float*)d_in[25];
  const float* rW3 = (const float*)d_in[26], *rb3 = (const float*)d_in[27];
  const float* tW1 = (const float*)d_in[28], *tb1 = (const float*)d_in[29];
  const float* tg1 = (const float*)d_in[30], *tbe1 = (const float*)d_in[31];
  const float* tW2 = (const float*)d_in[32], *tb2 = (const float*)d_in[33];
  const float* tg2 = (const float*)d_in[34], *tbe2 = (const float*)d_in[35];
  const float* tW3 = (const float*)d_in[36], *tb3 = (const float*)d_in[37];

  // workspace carve-out (~19 MB, 256B aligned)
  char* wp = (char*)d_ws;
  auto carve = [&](size_t bytes) -> void* {
    void* r = (void*)wp;
    wp += (bytes + 255) & ~(size_t)255;
    return r;
  };
  int*    idx    = (int*)   carve((size_t)B * N * KNN * 4);
  float*  act1   = (float*) carve((size_t)B * 64  * N * 4);
  float*  act2   = (float*) carve((size_t)B * 64  * N * 4);
  float*  act3   = (float*) carve((size_t)B * 128 * N * 4);
  float*  act4   = (float*) carve((size_t)B * 256 * N * 4);
  float*  pooled = (float*) carve((size_t)B * 1024 * 4);
  float*  hcat   = (float*) carve((size_t)B * 1152 * 4);
  float*  hd1    = (float*) carve((size_t)B * 512 * 4);
  float*  hd2    = (float*) carve((size_t)B * 256 * 4);
  half_t* W1h    = (half_t*)carve((size_t)64   * 32  * 2);
  half_t* W2h    = (half_t*)carve((size_t)64   * 128 * 2);
  half_t* W3h    = (half_t*)carve((size_t)128  * 128 * 2);
  half_t* W4h    = (half_t*)carve((size_t)256  * 256 * 2);
  half_t* W5h    = (half_t*)carve((size_t)1024 * 512 * 2);

  auto cvt = [&](const float* s, half_t* d, int O, int Cin, int Kpad) {
    int n = O * Kpad;
    cvt_w16<<<(n + 255) / 256, 256, 0, stream>>>(s, d, O, Cin, Kpad);
  };
  cvt(W1, W1h, 64, 18, 32);
  cvt(W2, W2h, 64, 128, 128);
  cvt(W3, W3h, 128, 128, 128);
  cvt(W4, W4h, 256, 256, 256);
  cvt(W5, W5h, 1024, 512, 512);

  dim3 knnGrid(N / 64, B);
  auto edge = [&](const float* in, const half_t* Wh, const float* g,
                  const float* be, float* out, int C, int O, int Kpad, int bn9) {
    dim3 grid(N / 16, (O + 63) / 64, B);
    edgeconv_wmma<<<grid, 128, 0, stream>>>(in, Wh, g, be, idx, out,
                                            B, C, N, O, Kpad, bn9);
  };

  knn_topk<<<knnGrid, 64, 0, stream>>>(x, idx, B, 9, N, /*bn9=*/1);
  edge(x,    W1h, g1, be1, act1,   9,  64,  32, 1);
  knn_topk<<<knnGrid, 64, 0, stream>>>(act1, idx, B, 64, N, 0);
  edge(act1, W2h, g2, be2, act2,  64,  64, 128, 0);
  knn_topk<<<knnGrid, 64, 0, stream>>>(act2, idx, B, 64, N, 0);
  edge(act2, W3h, g3, be3, act3,  64, 128, 128, 0);
  knn_topk<<<knnGrid, 64, 0, stream>>>(act3, idx, B, 128, N, 0);
  edge(act3, W4h, g4, be4, act4, 128, 256, 256, 0);

  fill_neg<<<(B * 1024 + 255) / 256, 256, 0, stream>>>(pooled, B * 1024);
  conv5_pool_wmma<<<dim3(N / 16, 16, B), 128, 0, stream>>>(
      act1, act2, act3, act4, W5h, g5, be5, pooled, B, N);

  concat_emb<<<(B * 1152 + 255) / 256, 256, 0, stream>>>(pooled, emb, obj, hcat, B);

  float* out_f = (float*)d_out;
  // rot head -> d_out[0:48]
  dense<<<(B * 512 + 255) / 256, 256, 0, stream>>>(hcat, rW1, rb1, rg1, rbe1, hd1, B, 1152, 512, 1);
  dense<<<(B * 256 + 255) / 256, 256, 0, stream>>>(hd1,  rW2, rb2, rg2, rbe2, hd2, B, 512, 256, 1);
  dense<<<(B * 6   + 255) / 256, 256, 0, stream>>>(hd2,  rW3, rb3, rg1, rbe1, out_f, B, 256, 6, 0);
  // trans head -> d_out[48:72]
  dense<<<(B * 512 + 255) / 256, 256, 0, stream>>>(hcat, tW1, tb1, tg1, tbe1, hd1, B, 1152, 512, 1);
  dense<<<(B * 256 + 255) / 256, 256, 0, stream>>>(hd1,  tW2, tb2, tg2, tbe2, hd2, B, 512, 256, 1);
  dense<<<(B * 3   + 255) / 256, 256, 0, stream>>>(hd2,  tW3, tb3, tg1, tbe1, out_f + 48, B, 256, 3, 0);
}
